// LocationSensitiveAttentionLayer_21234318312149
// MI455X (gfx1250) — matrix operand order repaired and checked by reading, not compiled, not run on hardware
//
#include <hip/hip_runtime.h>
#include <cstdint>
#include <cstddef>

// ---------------------------------------------------------------------------
// Location-sensitive attention for MI455X (gfx1250, wave32, WMMA).
// B=32, T_ENC=1024, T_DEC=512, E=512, D=1024, U=128, F=32, KW=31
// ---------------------------------------------------------------------------

#define B_    32
#define TENC  1024
#define TDEC  512
#define E_    512
#define D_    1024
#define U_    128
#define F_    32
#define KW_   31

typedef __attribute__((ext_vector_type(16))) __bf16 v16bf;
typedef __attribute__((ext_vector_type(8)))  float  v8f;

__device__ __forceinline__ __bf16 f2bf(float f) {
  unsigned u = __builtin_bit_cast(unsigned, f);
  unsigned r = u + 0x7FFFu + ((u >> 16) & 1u);     // round-to-nearest-even
  unsigned short h = (unsigned short)(r >> 16);
  return __builtin_bit_cast(__bf16, h);
}

// Convert a pair of f32 -> bf16 into fragment halves [idx], [idx+1].
// Uses v_cvt_pk_bf16_f32 when the builtin exists, else manual RNE.
__device__ __forceinline__ void cvt2(v16bf& a, int idx, float lo, float hi) {
#if __has_builtin(__builtin_amdgcn_cvt_pk_bf16_f32)
  auto r = __builtin_amdgcn_cvt_pk_bf16_f32(lo, hi);
  union { decltype(r) v; struct { __bf16 x, y; } s; } u;
  u.v = r;
  a[idx] = u.s.x;
  a[idx + 1] = u.s.y;
#else
  a[idx] = f2bf(lo);
  a[idx + 1] = f2bf(hi);
#endif
}

// A fragment (16x32, M x K) for v_wmma_f32_16x16x32_bf16, loaded from a
// row-major f32 matrix with on-the-fly bf16 conversion.
// Layout used here (contiguous k): lane L: M = L&15, half-group h = L>>4;
//   halves j = 0..15 -> K = 16h + j
__device__ __forceinline__ v16bf load_a_f32(const float* A, int lda, int m0,
                                            int k0, int lane) {
  const int tl = lane & 15, h = lane >> 4;
  const float* row = A + (size_t)(m0 + tl) * lda + k0;
  v16bf a;
#pragma unroll
  for (int j = 0; j < 16; j += 2) cvt2(a, j, row[16 * h + j], row[16 * h + j + 1]);
  return a;
}

// B fragment (32x16, K x N) from an N-major (transposed) bf16 matrix BT[N][K]:
// lane L holds N = L&15; halves j hold K = 16h + j  (contiguous 32B load).
__device__ __forceinline__ v16bf load_b_T(const __bf16* BT, int ldb, int n0,
                                          int k0, int lane) {
  const int tl = lane & 15, h = lane >> 4;
  const __bf16* p = BT + (size_t)(n0 + tl) * ldb + k0 + 16 * h;
  return *(const v16bf*)p;     // ldb % 16 == 0, k0 % 16 == 0 -> 32B aligned
}

// ---------------------------------------------------------------------------
// Prep: transpose weights to bf16, build M = conv_w @ W_location (K-padded),
// and fold b_a + conv_b@W_location into a keys bias.
// ---------------------------------------------------------------------------
__global__ void prep_weights(const float* __restrict__ Wm,   // (E,U)
                             const float* __restrict__ Wq,   // (D,U)
                             const float* __restrict__ Wl,   // (F,U)
                             const float* __restrict__ cw,   // (KW,1,F)
                             const float* __restrict__ cb,   // (F)
                             const float* __restrict__ ba,   // (U)
                             __bf16* __restrict__ WmT,       // (U,E)
                             __bf16* __restrict__ WqT,       // (U,D)
                             __bf16* __restrict__ MT,        // (U,32)
                             float*  __restrict__ kb) {      // (U)
  const int u = threadIdx.x;                  // 128 threads
  for (int e = 0; e < E_; ++e) WmT[u * E_ + e] = f2bf(Wm[(size_t)e * U_ + u]);
  for (int d = 0; d < D_; ++d) WqT[u * D_ + d] = f2bf(Wq[(size_t)d * U_ + u]);
  for (int k = 0; k < 32; ++k) {
    float s = 0.f;
    if (k < KW_)
      for (int f = 0; f < F_; ++f) s += cw[k * F_ + f] * Wl[f * U_ + u];
    MT[u * 32 + k] = f2bf(s);
  }
  float s = ba[u];
  for (int f = 0; f < F_; ++f) s += cb[f] * Wl[f * U_ + u];
  kb[u] = s;
}

// valuesT[b][e][t] = bf16(enc[b][t][e])  — LDS-tiled 32x32 transpose
__global__ void transpose_vals(const float* __restrict__ enc,
                               __bf16* __restrict__ valsT) {
  __shared__ __bf16 tile[32][33];
  int bid = blockIdx.x;
  const int eb = bid & 15;  bid >>= 4;        // E/32 = 16
  const int tb = bid & 31;                    // T/32 = 32
  const int b  = bid >> 5;
  const int tx = threadIdx.x & 31, ty = threadIdx.x >> 5;   // 32x8
  const float* src = enc + (size_t)b * TENC * E_;
#pragma unroll
  for (int i = 0; i < 32; i += 8) {
    int t = tb * 32 + ty + i, e = eb * 32 + tx;
    tile[ty + i][tx] = f2bf(src[(size_t)t * E_ + e]);
  }
  __syncthreads();
  __bf16* dst = valsT + (size_t)b * E_ * TENC;
#pragma unroll
  for (int i = 0; i < 32; i += 8) {
    int e = eb * 32 + ty + i, t = tb * 32 + tx;
    dst[(size_t)e * TENC + t] = tile[tx][ty + i];
  }
}

// ---------------------------------------------------------------------------
// keys = enc @ W_memory + (b_a + conv_b@W_loc).  One wave per 16-row tile;
// the wave keeps 8 accumulators (all of U) so the f32 A operand is loaded
// and converted exactly once.  8 independent WMMAs issue per k-step.
// ---------------------------------------------------------------------------
__global__ void gemm_keys(const float* __restrict__ enc,
                          const __bf16* __restrict__ WmT,
                          const float* __restrict__ kb,
                          float* __restrict__ keys) {
  const int lane = threadIdx.x & 31, w = threadIdx.x >> 5;
  const int g = blockIdx.x * 8 + w;           // 2048 row tiles
  const int m0 = g * 16;
  v8f acc[8];
#pragma unroll
  for (int ut = 0; ut < 8; ++ut) acc[ut] = (v8f){};
#pragma unroll 1
  for (int k0 = 0; k0 < E_; k0 += 32) {
    v16bf a = load_a_f32(enc, E_, m0, k0, lane);
#pragma unroll
    for (int ut = 0; ut < 8; ++ut) {
      v16bf bb = load_b_T(WmT, E_, ut * 16, k0, lane);
      acc[ut] = __builtin_amdgcn_wmma_f32_16x16x32_bf16(false, a, false, bb,
                                                        (short)0, acc[ut],
                                                        false, false);
    }
  }
  const int tl = lane & 15, h = lane >> 4;
#pragma unroll
  for (int ut = 0; ut < 8; ++ut) {
    const float kbv = kb[ut * 16 + tl];
#pragma unroll
    for (int r = 0; r < 8; ++r)
      keys[(size_t)(m0 + r + 8 * h) * U_ + ut * 16 + tl] = acc[ut][r] + kbv;
  }
}

// pq = dec @ W_query  (same structure, K = D = 1024)
__global__ void gemm_pq(const float* __restrict__ dec,
                        const __bf16* __restrict__ WqT,
                        float* __restrict__ pq) {
  const int lane = threadIdx.x & 31, w = threadIdx.x >> 5;
  const int g = blockIdx.x * 8 + w;           // 1024 row tiles
  const int m0 = g * 16;
  v8f acc[8];
#pragma unroll
  for (int ut = 0; ut < 8; ++ut) acc[ut] = (v8f){};
#pragma unroll 1
  for (int k0 = 0; k0 < D_; k0 += 32) {
    v16bf a = load_a_f32(dec, D_, m0, k0, lane);
#pragma unroll
    for (int ut = 0; ut < 8; ++ut) {
      v16bf bb = load_b_T(WqT, D_, ut * 16, k0, lane);
      acc[ut] = __builtin_amdgcn_wmma_f32_16x16x32_bf16(false, a, false, bb,
                                                        (short)0, acc[ut],
                                                        false, false);
    }
  }
  const int tl = lane & 15, h = lane >> 4;
#pragma unroll
  for (int ut = 0; ut < 8; ++ut)
#pragma unroll
    for (int r = 0; r < 8; ++r)
      pq[(size_t)(m0 + r + 8 * h) * U_ + ut * 16 + tl] = acc[ut][r];
}

// ---------------------------------------------------------------------------
// The sequential scan: one persistent 32-wave workgroup per batch.
// plf computed transposed:  Mt(16x32) x window(32x16)  via WMMA, where the
// B fragment is 16 *contiguous* floats of the LDS-resident padded cumulative
// alignment (31-tap "SAME" conv band, K padded 31->32).
// Energy partials go to epart[8][TENC] (one writer each; no LDS atomics).
// ---------------------------------------------------------------------------
__global__ void __launch_bounds__(1024) scan_kernel(
    const float* __restrict__ keys,     // (B*TENC, U) incl. bias
    const float* __restrict__ pq,       // (B*TDEC, U)
    const __bf16* __restrict__ MT,      // (U, 32)
    const float* __restrict__ va,       // (U)
    float* __restrict__ eout) {         // (B, TDEC, TENC)
  __shared__ float palign[15 + TENC + 32];   // padded cumulative alignment
  __shared__ float epart[8][TENC];           // per-u-tile energy partials
  __shared__ float red[32];
  __shared__ float pq_s[U_];
  __shared__ float va_s[U_];
  __shared__ float bcast[2];

  const int tid  = threadIdx.x;
  const int lane = tid & 31, w = tid >> 5;
  const int tl   = lane & 15, h = lane >> 4;
  const int b    = blockIdx.x;

  for (int i = tid; i < 15 + TENC + 32; i += 1024) palign[i] = 0.f;
  if (tid < U_) va_s[tid] = va[tid];

  // per-wave constant A fragment = M^T tile (u0..u0+15) x K(32)
  const int u0 = (w & 7) * 16;
  const int tg = w >> 3;                      // 4 groups of t-tiles
  v16bf amt;
  {
    const __bf16* row = MT + (u0 + tl) * 32;
#pragma unroll
    for (int j = 0; j < 16; ++j) amt[j] = row[16 * h + j];
  }
  __syncthreads();

  float va_r[8];
#pragma unroll
  for (int r = 0; r < 8; ++r) va_r[r] = va_s[u0 + 8 * h + r];

  const float* keysb = keys + (size_t)b * TENC * U_;

  for (int step = 0; step < TDEC; ++step) {
    if (tid < U_) pq_s[tid] = pq[((size_t)b * TDEC + step) * U_ + tid];
    __syncthreads();

    float pqr[8];
#pragma unroll
    for (int r = 0; r < 8; ++r) pqr[r] = pq_s[u0 + 8 * h + r];

#pragma unroll 1
    for (int i = 0; i < 16; ++i) {
      const int t0 = (tg * 16 + i) * 16;
      // B fragment: 16 consecutive padded-alignment values per lane
      const float* pw = palign + t0 + tl + 16 * h;
      v16bf bb;
#pragma unroll
      for (int j = 0; j < 16; j += 2) cvt2(bb, j, pw[j], pw[j + 1]);
      v8f c = {};
      c = __builtin_amdgcn_wmma_f32_16x16x32_bf16(false, amt, false, bb,
                                                  (short)0, c, false, false);
      // energy partial:  sum_u v_a[u] * tanh(keys + pq + plf)
      const int t = t0 + tl;
      const float* krow = keysb + (size_t)t * U_ + u0 + 8 * h;
      float p = 0.f;
#pragma unroll
      for (int r = 0; r < 8; ++r) {
        float x  = krow[r] + pqr[r] + c[r];
        float e2 = __expf(2.0f * x);
        float th = (e2 - 1.0f) * __builtin_amdgcn_rcpf(e2 + 1.0f);
        p += va_r[r] * th;
      }
      p += __shfl_xor(p, 16, 32);             // fold the two half-groups
      if (h == 0) epart[w & 7][t] = p;        // single writer per slot
    }
    __syncthreads();

    // --- softmax over TENC (1024 threads, 1 elem each) ---
    float x = 0.f;
#pragma unroll
    for (int pg = 0; pg < 8; ++pg) x += epart[pg][tid];
    float m = x;
#pragma unroll
    for (int off = 16; off; off >>= 1) m = fmaxf(m, __shfl_xor(m, off, 32));
    if (lane == 0) red[w] = m;
    __syncthreads();
    if (w == 0) {
      float mm = red[lane];
#pragma unroll
      for (int off = 16; off; off >>= 1) mm = fmaxf(mm, __shfl_xor(mm, off, 32));
      if (lane == 0) bcast[0] = mm;
    }
    __syncthreads();
    const float ex = __expf(x - bcast[0]);
    float s = ex;
#pragma unroll
    for (int off = 16; off; off >>= 1) s += __shfl_xor(s, off, 32);
    if (lane == 0) red[w] = s;
    __syncthreads();
    if (w == 0) {
      float ss = red[lane];
#pragma unroll
      for (int off = 16; off; off >>= 1) ss += __shfl_xor(ss, off, 32);
      if (lane == 0) bcast[1] = __builtin_amdgcn_rcpf(ss);
    }
    __syncthreads();
    const float a = ex * bcast[1];
    eout[((size_t)b * TDEC + step) * TENC + tid] = a;
    palign[15 + tid] += a;                    // cumulate_weights=True
    __syncthreads();
  }
}

// c_outputs[b][s][e] = e_out[b][s][:] @ values[b][:][e]   (bf16 WMMA)
// One wave per (s-tile, 8 e-tiles): e_out A-fragment reused 8x.
__global__ void gemm_cout(const float* __restrict__ eo,
                          const __bf16* __restrict__ valsT,
                          float* __restrict__ cout) {
  const int lane = threadIdx.x & 31, w = threadIdx.x >> 5;
  const int g = blockIdx.x * 8 + w;           // 32*32*4 = 4096 waves
  const int b = g >> 7;
  const int rem = g & 127;
  const int s0 = (rem >> 2) * 16;
  const int e0 = (rem & 3) * 128;             // 8 consecutive e-tiles
  const float*  A  = eo    + (size_t)b * TDEC * TENC;
  const __bf16* BT = valsT + (size_t)b * E_ * TENC;
  v8f acc[8];
#pragma unroll
  for (int j = 0; j < 8; ++j) acc[j] = (v8f){};
#pragma unroll 1
  for (int k0 = 0; k0 < TENC; k0 += 32) {
    v16bf a = load_a_f32(A, TENC, s0, k0, lane);
#pragma unroll
    for (int j = 0; j < 8; ++j) {
      v16bf bb = load_b_T(BT, TENC, e0 + j * 16, k0, lane);
      acc[j] = __builtin_amdgcn_wmma_f32_16x16x32_bf16(false, a, false, bb,
                                                       (short)0, acc[j],
                                                       false, false);
    }
  }
  const int tl = lane & 15, h = lane >> 4;
#pragma unroll
  for (int j = 0; j < 8; ++j)
#pragma unroll
    for (int r = 0; r < 8; ++r)
      cout[((size_t)b * TDEC + s0 + r + 8 * h) * E_ + e0 + j * 16 + tl] =
          acc[j][r];
}

// ---------------------------------------------------------------------------
extern "C" void kernel_launch(void* const* d_in, const int* in_sizes, int n_in,
                              void* d_out, int out_size, void* d_ws,
                              size_t ws_size, hipStream_t stream) {
  (void)in_sizes; (void)n_in; (void)out_size; (void)ws_size;
  const float* enc = (const float*)d_in[0];   // (B,TENC,E)
  const float* dec = (const float*)d_in[1];   // (B,TDEC,D)
  const float* Wm  = (const float*)d_in[2];   // (E,U)
  const float* Wq  = (const float*)d_in[3];   // (D,U)
  const float* Wl  = (const float*)d_in[4];   // (F,U)
  const float* cw  = (const float*)d_in[5];   // (KW,1,F)
  const float* cb  = (const float*)d_in[6];   // (F)
  const float* va  = (const float*)d_in[7];   // (U)
  const float* ba  = (const float*)d_in[8];   // (U)

  char* ws = (char*)d_ws;
  size_t off = 0;
  auto alloc = [&](size_t bytes) -> void* {
    void* p = ws + off;
    off = (off + bytes + 255) & ~(size_t)255;
    return p;
  };
  __bf16* valsT = (__bf16*)alloc((size_t)B_ * E_ * TENC * 2);   // 33.6 MB
  float*  keysW = (float*) alloc((size_t)B_ * TENC * U_ * 4);   // 16.8 MB
  float*  pqW   = (float*) alloc((size_t)B_ * TDEC * U_ * 4);   //  8.4 MB
  __bf16* WmT   = (__bf16*)alloc((size_t)U_ * E_ * 2);
  __bf16* WqT   = (__bf16*)alloc((size_t)U_ * D_ * 2);
  __bf16* MT    = (__bf16*)alloc((size_t)U_ * 32 * 2);
  float*  kb    = (float*) alloc((size_t)U_ * 4);

  float* coutP = (float*)d_out;                                  // (B,TDEC,E)
  float* eoutP = coutP + (size_t)B_ * TDEC * E_;                 // (B,TDEC,TENC)

  prep_weights<<<1, 128, 0, stream>>>(Wm, Wq, Wl, cw, cb, ba,
                                      WmT, WqT, MT, kb);
  transpose_vals<<<B_ * (TENC / 32) * (E_ / 32), 256, 0, stream>>>(enc, valsT);
  gemm_keys<<<(B_ * TENC / 16) / 8, 256, 0, stream>>>(enc, WmT, kb, keysW);
  gemm_pq<<<(B_ * TDEC / 16) / 8, 256, 0, stream>>>(dec, WqT, pqW);
  scan_kernel<<<B_, 1024, 0, stream>>>(keysW, pqW, MT, va, eoutP);
  gemm_cout<<<(B_ * TDEC / 16) * 4 / 8, 256, 0, stream>>>(eoutP, valsT, coutP);
}
